// FC_LSTM_18537078849957
// MI455X (gfx1250) — matrix operand order repaired and checked by reading, not compiled, not run on hardware
//
#include <hip/hip_runtime.h>
#include <hip/hip_bf16.h>

// ---------------- CDNA5 types ----------------
typedef __attribute__((ext_vector_type(16))) __bf16 v16bf;
typedef __attribute__((ext_vector_type(8)))  __bf16 v8bf;
typedef __attribute__((ext_vector_type(8)))  float  v8f;
typedef __attribute__((ext_vector_type(4)))  float  v4f;

#define BATCH  32
#define SEQLEN 2048
#define FDIM   128
#define HDIM   256
#define GDIM   1024            // 4*H, gate order i,f,o,g
#define OUT_T  (SEQLEN + 1)

// ---------------- workspace layout (bytes) ----------------
// xwb stored WMMA-fragment-major: [t][ntile 0..63][mtile 0..1][lane 0..31][8 f32]
#define XWB_OFF   0ull
#define XWB_BYTES ((size_t)SEQLEN * 64 * 2 * 32 * 8 * 4)
#define WXB_OFF   (XWB_OFF + XWB_BYTES)                   // bf16 [G][F]
#define WXB_BYTES ((size_t)GDIM * FDIM * 2)
#define WHB_OFF   (WXB_OFF + WXB_BYTES)                   // bf16 [G][H]
#define WHB_BYTES ((size_t)GDIM * HDIM * 2)
#define WCB_BYTES ((size_t)HDIM * HDIM * 2)
#define WCIB_OFF  (WHB_OFF + WHB_BYTES)
#define WCFB_OFF  (WCIB_OFF + WCB_BYTES)
#define WCOB_OFF  (WCFB_OFF + WCB_BYTES)

__device__ __forceinline__ v8f wmma_bf16(v16bf a, v16bf b, v8f c) {
  // (neg_a, A, neg_b, B, c_mod, C, reuse_a, reuse_b)
  return __builtin_amdgcn_wmma_f32_16x16x32_bf16(false, a, false, b, (short)0, c,
                                                 false, false);
}

__device__ __forceinline__ float sigf(float x) { return 1.f / (1.f + __expf(-x)); }

__device__ __forceinline__ v16bf cat8(v8bf lo, v8bf hi) {
  return __builtin_shufflevector(lo, hi, 0, 1, 2, 3, 4, 5, 6, 7,
                                         8, 9, 10, 11, 12, 13, 14, 15);
}

// ---------------- phase 0: f32 -> bf16 weight conversion ----------------
__global__ void cvt_f32_bf16(const float* __restrict__ src,
                             __bf16* __restrict__ dst, int n) {
  int i = blockIdx.x * blockDim.x + threadIdx.x;
  if (i < n) dst[i] = (__bf16)src[i];
}

// ---------------- phase 1: xwb = x@Wx^T + bx + bh (fragment-major) ----------------
// grid: 2048*16 blocks of 256 threads; 8 waves = 2 Mtiles x 4 Ntiles (64 cols/block)
__global__ __launch_bounds__(256) void xw_gemm(const float* __restrict__ x,
                                               const __bf16* __restrict__ Wxb,
                                               const float* __restrict__ bx,
                                               const float* __restrict__ bh,
                                               float* __restrict__ xwb) {
  const int t    = blockIdx.x >> 4;
  const int ng   = blockIdx.x & 15;
  const int w    = threadIdx.x >> 5;
  const int lane = threadIdx.x & 31;
  const int m    = w >> 2;           // batch tile 0..1
  const int nt   = w & 3;            // col tile within group 0..3
  const int ntile = ng * 4 + nt;     // global 16-col tile 0..63
  const int row  = lane & 15;
  const int half = lane >> 4;
  const int kb   = half * 16;
  const int col  = lane & 15;
  const int n    = ntile * 16 + col;

  const float*  xrow = x + ((size_t)(m * 16 + row) * SEQLEN + t) * FDIM;
  const __bf16* wrow = Wxb + (size_t)n * FDIM;

  v8f acc = (v8f)0.f;
#pragma unroll
  for (int kt = 0; kt < 4; ++kt) {
    // A frag: lane holds K = kt*32 + half*16 + 0..15 (contiguous, matches B)
    v4f p0 = *(const v4f*)(xrow + kt * 32 + kb);
    v4f p1 = *(const v4f*)(xrow + kt * 32 + kb + 4);
    v4f p2 = *(const v4f*)(xrow + kt * 32 + kb + 8);
    v4f p3 = *(const v4f*)(xrow + kt * 32 + kb + 12);
    v16bf a;
#pragma unroll
    for (int e = 0; e < 4; ++e) {
      a[e]      = (__bf16)p0[e];
      a[4 + e]  = (__bf16)p1[e];
      a[8 + e]  = (__bf16)p2[e];
      a[12 + e] = (__bf16)p3[e];
    }
    // B frag: lane holds N=col, K = kt*32 + half*16 + 0..15 (contiguous 32B)
    v16bf b = *(const v16bf*)(wrow + kt * 32 + half * 16);
    acc = wmma_bf16(a, b, acc);
  }
  const float bias = bx[n] + bh[n];
#pragma unroll
  for (int v = 0; v < 8; ++v) acc[v] += bias;
  // fragment-major store: one contiguous 32B per lane (1KB per wave-tile)
  float* of = xwb + ((((size_t)t * 64 + ntile) * 2 + m) * 32 + lane) * 8;
  *(v8f*)of = acc;
}

// ---------------- phase 2: persistent single-workgroup LSTM scan ----------------
// 1024 threads = 32 waves on one WGP. Wave (s,m): s=hidden slice (16 cols), m=batch tile.
__global__ __launch_bounds__(1024) void lstm_scan(const float* __restrict__ xwb,
                                                  const __bf16* __restrict__ Whb,
                                                  const __bf16* __restrict__ Wcib,
                                                  const __bf16* __restrict__ Wcfb,
                                                  const __bf16* __restrict__ Wcob,
                                                  const float* __restrict__ bci,
                                                  const float* __restrict__ bcf,
                                                  const float* __restrict__ bco,
                                                  float* __restrict__ out) {
  __shared__ __bf16 h_s[BATCH * HDIM];        // 16 KB
  __shared__ __bf16 c_s[2][BATCH * HDIM];     // 2 x 16 KB ping-pong

  const int tid  = threadIdx.x;
  const int w    = tid >> 5;
  const int lane = tid & 31;
  const int s    = w & 15;            // hidden slice
  const int m    = w >> 4;            // batch tile
  const int row  = lane & 15;
  const int half = lane >> 4;
  const int kb   = half * 16;
  const int col  = lane & 15;
  const int j    = s * 16 + col;      // this lane's hidden column (C/D N index)
  const int arow = m * 16 + row;      // this lane's A-frag batch row

  float* hid = out;
  float* mem = out + (size_t)OUT_T * BATCH * HDIM;

  // zero init: state + step-0 outputs
  for (int i = tid; i < BATCH * HDIM; i += 1024) {
    h_s[i]    = (__bf16)0.f;
    c_s[0][i] = (__bf16)0.f;
    c_s[1][i] = (__bf16)0.f;
    hid[i]    = 0.f;
    mem[i]    = 0.f;
  }
  const float bciv = bci[j], bcfv = bcf[j], bcov = bco[j];
  v8f creg = (v8f)0.f;                // private f32 cell tile (element v -> row m*16+v+half*8)

  // per-wave fragment-major xwb base: gate g lives at ntile = g*16 + s
  const size_t fragstep = (size_t)64 * 2 * 32 * 8;   // floats per timestep
  const float* xbase = xwb + (((size_t)s * 2 + m) * 32 + lane) * 8;
  const size_t goff = (size_t)16 * 2 * 32 * 8;       // +16 ntiles per gate

  for (int t = 0; t < SEQLEN; ++t) {
    __syncthreads();                  // h_s / c_s[rb] from previous step visible

    // ---- A-frags of h (reused across all 4 gate N-tiles)
    v16bf af[8];
#pragma unroll
    for (int kf = 0; kf < 8; ++kf) {
      v8bf lo = *(const v8bf*)(&h_s[arow * HDIM + kf * 32 + kb]);
      v8bf hi = *(const v8bf*)(&h_s[arow * HDIM + kf * 32 + 8 + kb]);
      af[kf] = cat8(lo, hi);
    }

    // ---- accumulators seeded from fragment-major xwb (2x b128 per gate)
    const float* xt = xbase + (size_t)t * fragstep;
    v8f accI = *(const v8f*)(xt + 0 * goff);
    v8f accF = *(const v8f*)(xt + 1 * goff);
    v8f accO = *(const v8f*)(xt + 2 * goff);
    v8f accG = *(const v8f*)(xt + 3 * goff);

    // ---- y += h @ Wh^T  (K=256 in 8 chained WMMAs, 4 gate tiles share A)
#pragma unroll
    for (int kf = 0; kf < 8; ++kf) {
      const int ko = kf * 32 + half * 16;
      v16bf bI = *(const v16bf*)(Whb + (size_t)(0 * HDIM + j) * HDIM + ko);
      v16bf bF = *(const v16bf*)(Whb + (size_t)(1 * HDIM + j) * HDIM + ko);
      v16bf bO = *(const v16bf*)(Whb + (size_t)(2 * HDIM + j) * HDIM + ko);
      v16bf bG = *(const v16bf*)(Whb + (size_t)(3 * HDIM + j) * HDIM + ko);
      accI = wmma_bf16(af[kf], bI, accI);
      accF = wmma_bf16(af[kf], bF, accF);
      accO = wmma_bf16(af[kf], bO, accO);
      accG = wmma_bf16(af[kf], bG, accG);
    }

    // ---- peephole i,f with c_old (read buffer rb)
    const __bf16* cold = c_s[(t + 1) & 1];
#pragma unroll
    for (int kf = 0; kf < 8; ++kf) {
      v8bf lo = *(const v8bf*)(&cold[arow * HDIM + kf * 32 + kb]);
      v8bf hi = *(const v8bf*)(&cold[arow * HDIM + kf * 32 + 8 + kb]);
      v16bf a = cat8(lo, hi);
      const int ko = kf * 32 + half * 16;
      v16bf bi = *(const v16bf*)(Wcib + (size_t)j * HDIM + ko);
      v16bf bf = *(const v16bf*)(Wcfb + (size_t)j * HDIM + ko);
      accI = wmma_bf16(a, bi, accI);
      accF = wmma_bf16(a, bf, accF);
    }

    // ---- gates + cell update (f32, exact element alignment with creg tile)
    v8f cnew;
#pragma unroll
    for (int v = 0; v < 8; ++v) {
      const float iv = sigf(accI[v] + bciv);
      const float fv = sigf(accF[v] + bcfv);
      const float gv = tanhf(accG[v]);
      cnew[v] = fv * creg[v] + iv * gv;
    }
    creg = cnew;

    // ---- publish c_new (write buffer wb) + memorys output
    __bf16* cw = c_s[t & 1];
#pragma unroll
    for (int v = 0; v < 8; ++v) {
      const int b = m * 16 + v + half * 8;
      cw[b * HDIM + j] = (__bf16)cnew[v];
      mem[((size_t)(t + 1) * BATCH + b) * HDIM + j] = cnew[v];
    }
    __syncthreads();                  // all c_new in LDS before o-peephole

    // ---- peephole o with c_new
#pragma unroll
    for (int kf = 0; kf < 8; ++kf) {
      v8bf lo = *(const v8bf*)(&cw[arow * HDIM + kf * 32 + kb]);
      v8bf hi = *(const v8bf*)(&cw[arow * HDIM + kf * 32 + 8 + kb]);
      v16bf a = cat8(lo, hi);
      v16bf bo = *(const v16bf*)(Wcob + (size_t)j * HDIM + kf * 32 + half * 16);
      accO = wmma_bf16(a, bo, accO);
    }

    // ---- h update + hiddens output
#pragma unroll
    for (int v = 0; v < 8; ++v) {
      const int b = m * 16 + v + half * 8;
      const float ov = sigf(accO[v] + bcov);
      const float hv = ov * tanhf(cnew[v]);
      h_s[b * HDIM + j] = (__bf16)hv;
      hid[((size_t)(t + 1) * BATCH + b) * HDIM + j] = hv;
    }
  }
}

// ---------------- launch ----------------
extern "C" void kernel_launch(void* const* d_in, const int* in_sizes, int n_in,
                              void* d_out, int out_size, void* d_ws, size_t ws_size,
                              hipStream_t stream) {
  const float* x   = (const float*)d_in[0];
  const float* Wx  = (const float*)d_in[1];
  const float* bx  = (const float*)d_in[2];
  const float* Wh  = (const float*)d_in[3];
  const float* bh  = (const float*)d_in[4];
  const float* Wci = (const float*)d_in[5];
  const float* bci = (const float*)d_in[6];
  const float* Wcf = (const float*)d_in[7];
  const float* bcf = (const float*)d_in[8];
  const float* Wco = (const float*)d_in[9];
  const float* bco = (const float*)d_in[10];

  char*   ws   = (char*)d_ws;
  float*  xwb  = (float*)(ws + XWB_OFF);
  __bf16* Wxb  = (__bf16*)(ws + WXB_OFF);
  __bf16* Whb  = (__bf16*)(ws + WHB_OFF);
  __bf16* Wcib = (__bf16*)(ws + WCIB_OFF);
  __bf16* Wcfb = (__bf16*)(ws + WCFB_OFF);
  __bf16* Wcob = (__bf16*)(ws + WCOB_OFF);

  // phase 0: weight conversion to bf16 (row-major [n][k], K contiguous)
  cvt_f32_bf16<<<(GDIM * FDIM + 255) / 256, 256, 0, stream>>>(Wx, Wxb, GDIM * FDIM);
  cvt_f32_bf16<<<(GDIM * HDIM + 255) / 256, 256, 0, stream>>>(Wh, Whb, GDIM * HDIM);
  cvt_f32_bf16<<<(HDIM * HDIM + 255) / 256, 256, 0, stream>>>(Wci, Wcib, HDIM * HDIM);
  cvt_f32_bf16<<<(HDIM * HDIM + 255) / 256, 256, 0, stream>>>(Wcf, Wcfb, HDIM * HDIM);
  cvt_f32_bf16<<<(HDIM * HDIM + 255) / 256, 256, 0, stream>>>(Wco, Wcob, HDIM * HDIM);

  // phase 1: whole-chip input projection (parallel over all timesteps)
  xw_gemm<<<SEQLEN * 16, 256, 0, stream>>>(x, Wxb, bx, bh, xwb);

  // phase 2: persistent single-workgroup recurrence (latency-optimized)
  lstm_scan<<<1, 1024, 0, stream>>>(xwb, Whb, Wcib, Wcfb, Wcob, bci, bcf, bco,
                                    (float*)d_out);
}
